// RnnInferenceNetwork_60816736911925
// MI455X (gfx1250) — matrix operand order repaired and checked by reading, not compiled, not run on hardware
//
#include <hip/hip_runtime.h>
#include <hip/hip_bf16.h>

// ---------------- problem constants (from reference setup_inputs) ----------
#define BS   128          // batch
#define TT   100          // timesteps
#define OBS  255          // obs dim
#define NS   20           // num_samples
#define BB   (BS*NS)      // 2560 total rows
#define EMB  256          // embedding dim
#define INP  262          // emb + 2*LAT
#define HID  512
#define G4   (4*HID)      // 2048
#define LAT  3
#define CHUNK 16          // rnn_states staging chunk (L2-resident, 84 MB)
#define LOG2PI_F 1.8378770664093453f

// output layout (flat concat, all f32)
#define O_Z   0                                  // [ns,bs,T,3]  -> b'*T*3 + t*3 + l
#define O_ENT ((size_t)BB*TT*LAT)                // 768000
#define O_LP  (O_ENT + (size_t)BB*TT)            // 1024000
#define O_RNN (O_LP + (size_t)BB*TT)             // 1280000, [b',h,t] -> b'*512*100 + h*100 + t

typedef __attribute__((ext_vector_type(16))) _Float16 v16h;
typedef __attribute__((ext_vector_type(8)))  _Float16 v8h;
typedef __attribute__((ext_vector_type(8)))  float    v8f;

// ---- WMMA fragment loaders (CDNA5 16x16x32 f16 layouts, wave32) ----------
// A (16x32, MxK): lane 0-15 row m; halfs 0..7 -> K = hs*8 + 0..7,
//                 halfs 8..15 -> K = 16 + hs*8 + 0..7  (hs = lane>>4)
__device__ __forceinline__ v16h load_a16x32(const _Float16* __restrict__ base,
                                            int ld, int row0, int k0, int lane) {
  int m = row0 + (lane & 15);
  const _Float16* p = base + (size_t)m * ld + k0 + ((lane >> 4) << 3);
  v8h lo = *(const v8h*)p;
  v8h hi = *(const v8h*)(p + 16);
  v16h a;
#pragma unroll
  for (int j = 0; j < 8; ++j) { a[j] = lo[j]; a[j + 8] = hi[j]; }
  return a;
}
// B (32x16, KxN): lane holds column n = lane&15; halfs 0..7 -> K = hs*8 + 0..7,
//                 halfs 8..15 -> K = 16 + hs*8 + 0..7  (hs = lane>>4)
__device__ __forceinline__ v16h load_b32x16(const _Float16* __restrict__ base,
                                            int ld, int col0, int k0, int lane) {
  int n = col0 + (lane & 15);
  const _Float16* p = base + (size_t)n * ld + k0 + ((lane >> 4) << 3);
  v8h lo = *(const v8h*)p;
  v8h hi = *(const v8h*)(p + 16);
  v16h b;
#pragma unroll
  for (int j = 0; j < 8; ++j) { b[j] = lo[j]; b[j + 8] = hi[j]; }
  return b;
}

__device__ __forceinline__ float sigf(float x) { return 1.0f / (1.0f + __expf(-x)); }

// ---------------- setup kernels -------------------------------------------
__global__ void k_tnorm(const float* __restrict__ times, float* __restrict__ tn) {
  int i = blockIdx.x * blockDim.x + threadIdx.x;
  if (i >= BS * TT) return;
  int b = i / TT;
  float t0 = times[b * TT], tN = times[b * TT + TT - 1];
  tn[i] = (times[i] - t0) / (tN - t0);
}

__global__ void k_buildX(const float* __restrict__ y, const float* __restrict__ tn,
                         _Float16* __restrict__ X) {
  int i = blockIdx.x * blockDim.x + threadIdx.x;
  if (i >= BS * TT * EMB) return;
  int row = i >> 8, c = i & 255;
  float v = (c < OBS) ? y[(size_t)row * OBS + c] : tn[row];
  X[i] = (_Float16)v;
}

__global__ void k_wembT(const float* __restrict__ W, _Float16* __restrict__ Wt) {
  int i = blockIdx.x * blockDim.x + threadIdx.x;
  if (i >= EMB * EMB) return;
  int n = i >> 8, k = i & 255;
  Wt[i] = (_Float16)W[k * EMB + n];     // Wt[n,k] = W_emb[k,n]
}

__global__ void k_wihe(const float* __restrict__ Wih, _Float16* __restrict__ We) {
  int i = blockIdx.x * blockDim.x + threadIdx.x;
  if (i >= G4 * EMB) return;
  int n = i >> 8, k = i & 255;
  We[i] = (_Float16)Wih[(size_t)n * INP + k];   // first 256 cols of W_ih
}

__global__ void k_whh16(const float* __restrict__ W, _Float16* __restrict__ Wh) {
  int i = blockIdx.x * blockDim.x + threadIdx.x;
  if (i >= G4 * HID) return;
  Wh[i] = (_Float16)W[i];
}

__global__ void k_wz3(const float* __restrict__ Wih, float* __restrict__ Wz) {
  int i = blockIdx.x * blockDim.x + threadIdx.x;
  if (i >= G4 * LAT) return;
  int n = i / LAT, j = i % LAT;
  // z_prev is z tiled x2 -> fold the two weight copies
  Wz[i] = Wih[(size_t)n * INP + 256 + j] + Wih[(size_t)n * INP + 259 + j];
}

// ---------------- GEMM 1: emb = X @ W_emb + b_emb  (M=12800,N=256,K=256) ---
__global__ void __launch_bounds__(256)
k_gemm_emb(const _Float16* __restrict__ X, const _Float16* __restrict__ Wt,
           const float* __restrict__ bemb, _Float16* __restrict__ out) {
  int lane = threadIdx.x & 31, wave = threadIdx.x >> 5;
  int m0 = blockIdx.x * 64 + (wave >> 1) * 16;
  int n0 = blockIdx.y * 32 + (wave & 1) * 16;
  v8f acc = {};
  for (int k0 = 0; k0 < EMB; k0 += 32) {
    v16h a = load_a16x32(X, EMB, m0, k0, lane);
    v16h b = load_b32x16(Wt, EMB, n0, k0, lane);
    acc = __builtin_amdgcn_wmma_f32_16x16x32_f16(false, a, false, b, (short)0, acc, false, false);
  }
  int n = n0 + (lane & 15);
  float bv = bemb[n];
  int mrow = m0 + ((lane >> 4) << 3);
#pragma unroll
  for (int s = 0; s < 8; ++s)
    out[(size_t)(mrow + s) * EMB + n] = (_Float16)(acc[s] + bv);
}

// ------ GEMM 2: P = emb @ W_ih[:, :256]^T + (b_ih+b_hh)  (M=12800,N=2048) --
__global__ void __launch_bounds__(256)
k_gemm_pre(const _Float16* __restrict__ E, const _Float16* __restrict__ We,
           const float* __restrict__ bih, const float* __restrict__ bhh,
           float* __restrict__ P) {
  int lane = threadIdx.x & 31, wave = threadIdx.x >> 5;
  int m0 = blockIdx.x * 64 + (wave >> 1) * 16;
  int n0 = blockIdx.y * 32 + (wave & 1) * 16;
  v8f acc = {};
  for (int k0 = 0; k0 < EMB; k0 += 32) {
    v16h a = load_a16x32(E, EMB, m0, k0, lane);
    v16h b = load_b32x16(We, EMB, n0, k0, lane);
    acc = __builtin_amdgcn_wmma_f32_16x16x32_f16(false, a, false, b, (short)0, acc, false, false);
  }
  int n = n0 + (lane & 15);
  float bv = bih[n] + bhh[n];
  int mrow = m0 + ((lane >> 4) << 3);
#pragma unroll
  for (int s = 0; s < 8; ++s)
    P[(size_t)(mrow + s) * G4 + n] = acc[s] + bv;
}

// -------- per-step fused LSTM cell: gates WMMA + activations + stores ------
// grid (BB/64=40, HID/32=16), 256 threads. N-tile indexes the HIDDEN unit;
// all 4 gate quadrants computed here so the cell epilogue is local.
// c_new goes to an L2-resident staging chunk [t&15][b'][h] with fully
// coalesced stores; a flush kernel later emits the [b',h,t] output layout
// with full-line DRAM writes (avoids ~4x write amplification on 524 MB).
__global__ void __launch_bounds__(256)
k_step(const _Float16* __restrict__ hin, const _Float16* __restrict__ Whh,
       const float* __restrict__ P, const float* __restrict__ Wz3,
       const float* __restrict__ z3, float* __restrict__ cst,
       _Float16* __restrict__ hout, float* __restrict__ stag, int t) {
  int lane = threadIdx.x & 31, wave = threadIdx.x >> 5;
  int m0 = blockIdx.x * 64 + (wave >> 1) * 16;
  int n0 = blockIdx.y * 32 + (wave & 1) * 16;
  v8f acc0 = {}, acc1 = {}, acc2 = {}, acc3 = {};
  for (int k0 = 0; k0 < HID; k0 += 32) {
    v16h a = load_a16x32(hin, HID, m0, k0, lane);          // one A for 4 WMMAs
    v16h b0 = load_b32x16(Whh, HID, 0 * HID + n0, k0, lane);
    acc0 = __builtin_amdgcn_wmma_f32_16x16x32_f16(false, a, false, b0, (short)0, acc0, false, false);
    v16h b1 = load_b32x16(Whh, HID, 1 * HID + n0, k0, lane);
    acc1 = __builtin_amdgcn_wmma_f32_16x16x32_f16(false, a, false, b1, (short)0, acc1, false, false);
    v16h b2 = load_b32x16(Whh, HID, 2 * HID + n0, k0, lane);
    acc2 = __builtin_amdgcn_wmma_f32_16x16x32_f16(false, a, false, b2, (short)0, acc2, false, false);
    v16h b3 = load_b32x16(Whh, HID, 3 * HID + n0, k0, lane);
    acc3 = __builtin_amdgcn_wmma_f32_16x16x32_f16(false, a, false, b3, (short)0, acc3, false, false);
  }
  int n = n0 + (lane & 15);
  float wz[4][3];
#pragma unroll
  for (int q = 0; q < 4; ++q)
#pragma unroll
    for (int j = 0; j < 3; ++j) wz[q][j] = Wz3[(size_t)(q * HID + n) * 3 + j];
  int mbase = m0 + ((lane >> 4) << 3);
  float* stagp = stag + (size_t)(t & (CHUNK - 1)) * ((size_t)BB * HID);
#pragma unroll
  for (int s = 0; s < 8; ++s) {
    int m = mbase + s;                       // row b' in [0, BB)
    int borig = m & (BS - 1);                // b' % bs
    const float* prow = P + ((size_t)borig * TT + t) * G4;
    float za = z3[m * 3 + 0], zb = z3[m * 3 + 1], zc = z3[m * 3 + 2];
    float gi = acc0[s] + prow[0 * HID + n] + za * wz[0][0] + zb * wz[0][1] + zc * wz[0][2];
    float gf = acc1[s] + prow[1 * HID + n] + za * wz[1][0] + zb * wz[1][1] + zc * wz[1][2];
    float gg = acc2[s] + prow[2 * HID + n] + za * wz[2][0] + zb * wz[2][1] + zc * wz[2][2];
    float go = acc3[s] + prow[3 * HID + n] + za * wz[3][0] + zb * wz[3][1] + zc * wz[3][2];
    float iv = sigf(gi), fv = sigf(gf), gv = tanhf(gg), ov = sigf(go);
    size_t idx = (size_t)m * HID + n;
    float cn = fv * cst[idx] + iv * gv;
    cst[idx] = cn;
    hout[idx] = (_Float16)(ov * tanhf(cn));
    stagp[idx] = cn;                         // coalesced (lanes -> consecutive h)
  }
}

// ---- flush staging chunk -> rnn_states output [b', h, t] ------------------
// Scattered 4B reads hit the L2-resident 84MB chunk; DRAM writes are
// aligned float4 runs along contiguous t (full-line writes, written once).
__global__ void __launch_bounds__(256)
k_flush(const float* __restrict__ stag, float* __restrict__ out_rnn,
        int t0, int cl) {
  int i = blockIdx.x * blockDim.x + threadIdx.x;   // i = m*HID + n
  if (i >= BB * HID) return;
  int m = i >> 9, n = i & (HID - 1);
  float* op = out_rnn + (size_t)m * HID * TT + (size_t)n * TT + t0;
  const float* sp = stag + i;
  const size_t plane = (size_t)BB * HID;
  if (cl == CHUNK) {
#pragma unroll
    for (int q = 0; q < 4; ++q) {
      float4 v;
      v.x = sp[(size_t)(4 * q + 0) * plane];
      v.y = sp[(size_t)(4 * q + 1) * plane];
      v.z = sp[(size_t)(4 * q + 2) * plane];
      v.w = sp[(size_t)(4 * q + 3) * plane];
      *(float4*)(op + 4 * q) = v;                  // 16B-aligned (n*100+t0 ≡ 0 mod 4)
    }
  } else {
    float4 v;                                       // tail chunk (cl == 4)
    v.x = sp[0 * plane]; v.y = sp[1 * plane];
    v.z = sp[2 * plane]; v.w = sp[3 * plane];
    *(float4*)op = v;
  }
}

// ------- per-step latent head: 512->3 GEMVs, softplus, rsample, logp/ent ---
__global__ void __launch_bounds__(256)
k_latent(const _Float16* __restrict__ h, const float* __restrict__ eps,
         const float* __restrict__ Wm, const float* __restrict__ bm,
         const float* __restrict__ Ws, const float* __restrict__ bs_,
         float* __restrict__ z3, float* __restrict__ out, int t) {
  int lane = threadIdx.x & 31;
  int row = blockIdx.x * 8 + (threadIdx.x >> 5);     // one wave per row, BB=320*8
  float sm0 = 0, sm1 = 0, sm2 = 0, ss0 = 0, ss1 = 0, ss2 = 0;
  const _Float16* hp = h + (size_t)row * HID + lane * 16;
#pragma unroll
  for (int j = 0; j < 16; ++j) {
    float hv = (float)hp[j];
    int k = lane * 16 + j;
    sm0 += hv * Wm[k * 3 + 0]; sm1 += hv * Wm[k * 3 + 1]; sm2 += hv * Wm[k * 3 + 2];
    ss0 += hv * Ws[k * 3 + 0]; ss1 += hv * Ws[k * 3 + 1]; ss2 += hv * Ws[k * 3 + 2];
  }
#pragma unroll
  for (int off = 16; off > 0; off >>= 1) {
    sm0 += __shfl_xor(sm0, off, 32); sm1 += __shfl_xor(sm1, off, 32);
    sm2 += __shfl_xor(sm2, off, 32); ss0 += __shfl_xor(ss0, off, 32);
    ss1 += __shfl_xor(ss1, off, 32); ss2 += __shfl_xor(ss2, off, 32);
  }
  if (lane == 0) {
    float sm[3] = {sm0, sm1, sm2}, ss[3] = {ss0, ss1, ss2};
    float logp = 0.f, ent = 0.f;
#pragma unroll
    for (int l = 0; l < 3; ++l) {
      float mean = sm[l] + bm[l];
      float sr = ss[l] + bs_[l];
      float sp = (sr > 20.f) ? sr : log1pf(__expf(sr));   // softplus
      float sd = sp + 1e-4f;
      float e = eps[((size_t)t * BB + row) * 3 + l];
      float zv = mean + sd * e;
      z3[row * 3 + l] = zv;
      out[O_Z + (size_t)row * TT * 3 + (size_t)t * 3 + l] = zv;
      float ls = logf(sd);
      logp += -0.5f * e * e - ls - 0.5f * LOG2PI_F;
      ent  += 0.5f + 0.5f * LOG2PI_F + ls;
    }
    out[O_ENT + (size_t)row * TT + t] = ent;
    out[O_LP  + (size_t)row * TT + t] = logp;
  }
}

// ---------------------------------------------------------------------------
extern "C" void kernel_launch(void* const* d_in, const int* in_sizes, int n_in,
                              void* d_out, int out_size, void* d_ws, size_t ws_size,
                              hipStream_t stream) {
  const float* y     = (const float*)d_in[0];
  const float* times = (const float*)d_in[1];
  const float* eps   = (const float*)d_in[2];
  const float* W_emb = (const float*)d_in[3];
  const float* b_emb = (const float*)d_in[4];
  const float* W_ih  = (const float*)d_in[5];
  const float* W_hh  = (const float*)d_in[6];
  const float* b_ih  = (const float*)d_in[7];
  const float* b_hh  = (const float*)d_in[8];
  const float* W_m   = (const float*)d_in[9];
  const float* b_m   = (const float*)d_in[10];
  const float* W_s   = (const float*)d_in[11];
  const float* b_s   = (const float*)d_in[12];
  float* out = (float*)d_out;

  // workspace carve-out (~210 MB), 256B-aligned slabs
  char* ws = (char*)d_ws;
  size_t off = 0;
  auto carve = [&](size_t bytes) -> char* {
    char* p = ws + off;
    off = (off + bytes + 255) & ~(size_t)255;
    return p;
  };
  float*    tn     = (float*)   carve((size_t)BS * TT * 4);
  _Float16* X16    = (_Float16*)carve((size_t)BS * TT * EMB * 2);
  _Float16* WembT  = (_Float16*)carve((size_t)EMB * EMB * 2);
  _Float16* emb16  = (_Float16*)carve((size_t)BS * TT * EMB * 2);
  _Float16* Wihe   = (_Float16*)carve((size_t)G4 * EMB * 2);
  _Float16* Whh16  = (_Float16*)carve((size_t)G4 * HID * 2);
  float*    Wz3    = (float*)   carve((size_t)G4 * LAT * 4);
  float*    P      = (float*)   carve((size_t)BS * TT * G4 * 4);         // 104.9 MB
  float*    cst    = (float*)   carve((size_t)BB * HID * 4);
  _Float16* h16a   = (_Float16*)carve((size_t)BB * HID * 2);
  _Float16* h16b   = (_Float16*)carve((size_t)BB * HID * 2);
  float*    z3     = (float*)   carve((size_t)BB * LAT * 4);
  float*    stag   = (float*)   carve((size_t)CHUNK * BB * HID * 4);     // 83.9 MB

  // ---- setup / precompute ----
  k_tnorm <<<(BS * TT + 255) / 256, 256, 0, stream>>>(times, tn);
  k_buildX<<<(BS * TT * EMB + 255) / 256, 256, 0, stream>>>(y, tn, X16);
  k_wembT <<<(EMB * EMB + 255) / 256, 256, 0, stream>>>(W_emb, WembT);
  k_wihe  <<<(G4 * EMB + 255) / 256, 256, 0, stream>>>(W_ih, Wihe);
  k_whh16 <<<(G4 * HID + 255) / 256, 256, 0, stream>>>(W_hh, Whh16);
  k_wz3   <<<(G4 * LAT + 255) / 256, 256, 0, stream>>>(W_ih, Wz3);
  hipMemsetAsync(cst, 0, (size_t)BB * HID * 4, stream);
  hipMemsetAsync(h16a, 0, (size_t)BB * HID * 2, stream);
  hipMemsetAsync(h16b, 0, (size_t)BB * HID * 2, stream);
  hipMemsetAsync(z3, 0, (size_t)BB * LAT * 4, stream);

  k_gemm_emb<<<dim3(BS * TT / 64, EMB / 32), 256, 0, stream>>>(X16, WembT, b_emb, emb16);
  k_gemm_pre<<<dim3(BS * TT / 64, G4 / 32), 256, 0, stream>>>(emb16, Wihe, b_ih, b_hh, P);

  // ---- sequential LSTM over time, h double-buffered by pointer swap ----
  _Float16* hbuf[2] = {h16a, h16b};
  for (int t = 0; t < TT; ++t) {
    _Float16* hin  = hbuf[t & 1];
    _Float16* hout = hbuf[(t + 1) & 1];
    k_step  <<<dim3(BB / 64, HID / 32), 256, 0, stream>>>(
        hin, Whh16, P, Wz3, z3, cst, hout, stag, t);
    k_latent<<<BB / 8, 256, 0, stream>>>(
        hout, eps, W_m, b_m, W_s, b_s, z3, out, t);
    if ((t & (CHUNK - 1)) == (CHUNK - 1) || t == TT - 1) {
      int t0 = t & ~(CHUNK - 1);
      k_flush<<<(BB * HID + 255) / 256, 256, 0, stream>>>(
          stag, out + O_RNN, t0, t - t0 + 1);
    }
  }
}
